// MemoryOptimizedAttention_56788057587825
// MI455X (gfx1250) — matrix operand order repaired and checked by reading, not compiled, not run on hardware
//
#include <hip/hip_runtime.h>
#include <hip/hip_bf16.h>
#include <math.h>

// ---------------------------------------------------------------------------
// MHA forward for B=4, T=2048, D=1024, H=16, DH=64 on gfx1250 (MI455X).
// bf16 WMMA (v_wmma_f32_16x16x32_bf16) everywhere, f32 accumulation.
//  - GEMMs: 32x64 per wave; K-loop unrolled x2 as X/Y ping-pong (no register
//    rotation moves, loads overlap WMMAs).
//  - Attention: flash-style, 64-wide KV blocks; V fragments issued before the
//    softmax VALU work; LDS P relayout with conflict-free padding and
//    wave-local s_wait_dscnt sync.
// ---------------------------------------------------------------------------

typedef __bf16 bf16_t;
typedef __attribute__((ext_vector_type(16))) __bf16 v16bf;
typedef __attribute__((ext_vector_type(8)))  float  v8f;

#define B_   4
#define T_   2048
#define D_   1024
#define H_   16
#define DH_  64
#define SCALE_ 0.125f   // 1/sqrt(64)

__device__ __forceinline__ v8f wmma_bf16(v16bf a, v16bf b, v8f c) {
  // (neg_a, A, neg_b, B, c_mod, C, reuse_a, reuse_b)
  return __builtin_amdgcn_wmma_f32_16x16x32_bf16(false, a, false, b, (short)0, c,
                                                 false, false);
}

union FragU { v16bf v; uint4 u[2]; };

// A-matrix 16x32 bf16 fragment (ISA 7.12.2): lane<16 -> M=lane, K = e (e<8) or
// e+8 (e>=8); lane>=16 -> M=lane-16, same +8.  Two contiguous 16B runs/lane.
__device__ __forceinline__ v16bf load_a_frag(const bf16_t* __restrict__ base,
                                             int row_stride, int row, int koff,
                                             int laneHi) {
  FragU f;
  const bf16_t* p = base + (size_t)row * row_stride + koff + (laneHi ? 8 : 0);
  f.u[0] = *(const uint4*)(p);
  f.u[1] = *(const uint4*)(p + 16);
  return f.v;
}

// B-matrix 32x16 bf16 fragment: lane<16 -> N=lane, K = e (e<8) or e+8 (e>=8);
// lane>=16 -> N=lane-16, same +8.  Two contiguous 16B runs/lane.
__device__ __forceinline__ v16bf load_b_frag(const bf16_t* __restrict__ base,
                                             int col_stride, int col, int koff,
                                             int laneHi) {
  FragU f;
  const bf16_t* p = base + (size_t)col * col_stride + koff + (laneHi ? 8 : 0);
  f.u[0] = *(const uint4*)(p);
  f.u[1] = *(const uint4*)(p + 16);
  return f.v;
}

// ---------------------------------------------------------------------------
__global__ void cvt_f32_bf16_kernel(const float* __restrict__ src,
                                    bf16_t* __restrict__ dst, int n) {
  int i = blockIdx.x * blockDim.x + threadIdx.x;
  int stride = gridDim.x * blockDim.x;
  for (; i < n; i += stride) dst[i] = (bf16_t)src[i];
}

// ---------------------------------------------------------------------------
// C[M,N] = A[M,K] @ W[N,K]^T + bias.   A,W bf16 row-major.
// Block = 128 threads (4 waves, 2x2), block tile 64x128, wave tile 32x64.
// K-loop unrolled by 64 with X/Y fragment ping-pong, last iteration peeled.
// mode: 1 -> bf16 out to [B,H,T,DH]; 2 -> bf16 out to [B,H,DH,T];
//       3 -> f32 row-major out.   Requires Kdim % 64 == 0.
__global__ __launch_bounds__(128)
void gemm_xwT_kernel(const bf16_t* __restrict__ A, const bf16_t* __restrict__ W,
                     const float* __restrict__ bias,
                     bf16_t* __restrict__ out_bf, float* __restrict__ out_f32,
                     int M, int N, int Kdim, int mode) {
  const int lane   = threadIdx.x & 31;
  const int wave   = threadIdx.x >> 5;
  const int laneHi = lane >> 4;
  const int ln     = lane & 15;
  const int m0 = blockIdx.x * 64 + (wave & 1) * 32;
  const int n0 = blockIdx.y * 128 + (wave >> 1) * 64;

  v8f c[2][4];
  for (int v = 0; v < 4; ++v) {
    float bv = bias[n0 + v * 16 + ln];
    for (int g = 0; g < 2; ++g)
      for (int r = 0; r < 8; ++r) c[g][v][r] = bv;
  }

  const int ra0 = m0 + ln, ra1 = m0 + 16 + ln;
  const int cb0 = n0 + ln, cb1 = n0 + 16 + ln, cb2 = n0 + 32 + ln,
            cb3 = n0 + 48 + ln;

  // X set (k), Y set (k+32)
  v16bf xa0 = load_a_frag(A, Kdim, ra0, 0, laneHi);
  v16bf xa1 = load_a_frag(A, Kdim, ra1, 0, laneHi);
  v16bf xb0 = load_b_frag(W, Kdim, cb0, 0, laneHi);
  v16bf xb1 = load_b_frag(W, Kdim, cb1, 0, laneHi);
  v16bf xb2 = load_b_frag(W, Kdim, cb2, 0, laneHi);
  v16bf xb3 = load_b_frag(W, Kdim, cb3, 0, laneHi);
  v16bf ya0, ya1, yb0, yb1, yb2, yb3;

  int k0 = 0;
  for (; k0 < Kdim - 64; k0 += 64) {
    ya0 = load_a_frag(A, Kdim, ra0, k0 + 32, laneHi);
    ya1 = load_a_frag(A, Kdim, ra1, k0 + 32, laneHi);
    yb0 = load_b_frag(W, Kdim, cb0, k0 + 32, laneHi);
    yb1 = load_b_frag(W, Kdim, cb1, k0 + 32, laneHi);
    yb2 = load_b_frag(W, Kdim, cb2, k0 + 32, laneHi);
    yb3 = load_b_frag(W, Kdim, cb3, k0 + 32, laneHi);
    c[0][0] = wmma_bf16(xa0, xb0, c[0][0]);
    c[1][0] = wmma_bf16(xa1, xb0, c[1][0]);
    c[0][1] = wmma_bf16(xa0, xb1, c[0][1]);
    c[1][1] = wmma_bf16(xa1, xb1, c[1][1]);
    c[0][2] = wmma_bf16(xa0, xb2, c[0][2]);
    c[1][2] = wmma_bf16(xa1, xb2, c[1][2]);
    c[0][3] = wmma_bf16(xa0, xb3, c[0][3]);
    c[1][3] = wmma_bf16(xa1, xb3, c[1][3]);
    xa0 = load_a_frag(A, Kdim, ra0, k0 + 64, laneHi);
    xa1 = load_a_frag(A, Kdim, ra1, k0 + 64, laneHi);
    xb0 = load_b_frag(W, Kdim, cb0, k0 + 64, laneHi);
    xb1 = load_b_frag(W, Kdim, cb1, k0 + 64, laneHi);
    xb2 = load_b_frag(W, Kdim, cb2, k0 + 64, laneHi);
    xb3 = load_b_frag(W, Kdim, cb3, k0 + 64, laneHi);
    c[0][0] = wmma_bf16(ya0, yb0, c[0][0]);
    c[1][0] = wmma_bf16(ya1, yb0, c[1][0]);
    c[0][1] = wmma_bf16(ya0, yb1, c[0][1]);
    c[1][1] = wmma_bf16(ya1, yb1, c[1][1]);
    c[0][2] = wmma_bf16(ya0, yb2, c[0][2]);
    c[1][2] = wmma_bf16(ya1, yb2, c[1][2]);
    c[0][3] = wmma_bf16(ya0, yb3, c[0][3]);
    c[1][3] = wmma_bf16(ya1, yb3, c[1][3]);
  }
  // peeled tail: last 64 of K, no further prefetch
  ya0 = load_a_frag(A, Kdim, ra0, k0 + 32, laneHi);
  ya1 = load_a_frag(A, Kdim, ra1, k0 + 32, laneHi);
  yb0 = load_b_frag(W, Kdim, cb0, k0 + 32, laneHi);
  yb1 = load_b_frag(W, Kdim, cb1, k0 + 32, laneHi);
  yb2 = load_b_frag(W, Kdim, cb2, k0 + 32, laneHi);
  yb3 = load_b_frag(W, Kdim, cb3, k0 + 32, laneHi);
  c[0][0] = wmma_bf16(xa0, xb0, c[0][0]);
  c[1][0] = wmma_bf16(xa1, xb0, c[1][0]);
  c[0][1] = wmma_bf16(xa0, xb1, c[0][1]);
  c[1][1] = wmma_bf16(xa1, xb1, c[1][1]);
  c[0][2] = wmma_bf16(xa0, xb2, c[0][2]);
  c[1][2] = wmma_bf16(xa1, xb2, c[1][2]);
  c[0][3] = wmma_bf16(xa0, xb3, c[0][3]);
  c[1][3] = wmma_bf16(xa1, xb3, c[1][3]);
  c[0][0] = wmma_bf16(ya0, yb0, c[0][0]);
  c[1][0] = wmma_bf16(ya1, yb0, c[1][0]);
  c[0][1] = wmma_bf16(ya0, yb1, c[0][1]);
  c[1][1] = wmma_bf16(ya1, yb1, c[1][1]);
  c[0][2] = wmma_bf16(ya0, yb2, c[0][2]);
  c[1][2] = wmma_bf16(ya1, yb2, c[1][2]);
  c[0][3] = wmma_bf16(ya0, yb3, c[0][3]);
  c[1][3] = wmma_bf16(ya1, yb3, c[1][3]);

  for (int g = 0; g < 2; ++g)
    for (int v = 0; v < 4; ++v)
      for (int r = 0; r < 8; ++r) {
        int m = m0 + g * 16 + r + laneHi * 8;  // C: VGPR r -> M=r (+8 hi half)
        int n = n0 + v * 16 + ln;              // lane -> N
        float val = c[g][v][r];
        if (mode == 3) {
          out_f32[(size_t)m * N + n] = val;
        } else {
          int bb = m / T_, t = m % T_;
          int h  = n / DH_, d = n % DH_;
          if (mode == 1)       // [B,H,T,DH]
            out_bf[(((size_t)(bb * H_ + h) * T_ + t) * DH_) + d] = (bf16_t)val;
          else                 // [B,H,DH,T]
            out_bf[(((size_t)(bb * H_ + h) * DH_ + d) * T_) + t] = (bf16_t)val;
        }
      }
}

// ---------------------------------------------------------------------------
// Flash attention.  Grid: (T/64, H, B).  4 waves, wave w owns 16 query rows,
// KV processed in 64-column blocks.  Qh,Kh: [B,H,T,DH];  Vt: [B,H,DH,T];
// O: [B,T,D] bf16.  LDS P staging is wave-private: 72-element row padding
// makes the 4x ds_load_b128 per lane hit all 64 banks exactly once per half.
__global__ __launch_bounds__(128)
void attn_kernel(const bf16_t* __restrict__ Qh, const bf16_t* __restrict__ Kh,
                 const bf16_t* __restrict__ Vt, bf16_t* __restrict__ O) {
  __shared__ bf16_t plds[4][16][72];

  const int lane   = threadIdx.x & 31;
  const int wave   = threadIdx.x >> 5;
  const int laneHi = lane >> 4;
  const int ln     = lane & 15;
  const int h  = blockIdx.y;
  const int b  = blockIdx.z;
  const int t0 = blockIdx.x * 64 + wave * 16;

  const bf16_t* Qbase = Qh + (size_t)(b * H_ + h) * T_ * DH_;
  const bf16_t* Kbase = Kh + (size_t)(b * H_ + h) * T_ * DH_;
  const bf16_t* Vbase = Vt + (size_t)(b * H_ + h) * DH_ * T_;

  // Q fragments for this wave's 16 rows, K-dim 0..63 (held for whole loop).
  v16bf aq0 = load_a_frag(Qbase, DH_, t0 + ln, 0, laneHi);
  v16bf aq1 = load_a_frag(Qbase, DH_, t0 + ln, 32, laneHi);

  float mrow[8], lrow[8];
  v8f acc[4];
  for (int r = 0; r < 8; ++r) { mrow[r] = -1e30f; lrow[r] = 0.f; }
  for (int v = 0; v < 4; ++v)
    for (int r = 0; r < 8; ++r) acc[v][r] = 0.f;

  for (int tk = 0; tk < T_; tk += 64) {
    // ---- load all 8 K fragments, then run the 8 score WMMAs ----
    v16bf kb[4][2];
#pragma unroll
    for (int j = 0; j < 4; ++j) {
      kb[j][0] = load_b_frag(Kbase, DH_, tk + j * 16 + ln, 0, laneHi);
      kb[j][1] = load_b_frag(Kbase, DH_, tk + j * 16 + ln, 32, laneHi);
    }
    float p[4][8];
#pragma unroll
    for (int j = 0; j < 4; ++j) {
      v8f s = {};
      s = wmma_bf16(aq0, kb[j][0], s);
      s = wmma_bf16(aq1, kb[j][1], s);
#pragma unroll
      for (int r = 0; r < 8; ++r) p[j][r] = s[r] * SCALE_;
    }

    // ---- issue V fragments now; they fly under the softmax VALU work ----
    v16bf bv[4][2];
#pragma unroll
    for (int v = 0; v < 4; ++v) {
      bv[v][0] = load_b_frag(Vbase, T_, v * 16 + ln, tk,      laneHi);
      bv[v][1] = load_b_frag(Vbase, T_, v * 16 + ln, tk + 32, laneHi);
    }

    // ---- online softmax (row reductions across each 16-lane half) ----
    float alpha[8];
#pragma unroll
    for (int r = 0; r < 8; ++r) {
      float x = fmaxf(fmaxf(p[0][r], p[1][r]), fmaxf(p[2][r], p[3][r]));
      for (int off = 1; off < 16; off <<= 1) x = fmaxf(x, __shfl_xor(x, off, 32));
      float mnew = fmaxf(mrow[r], x);
      alpha[r] = __expf(mrow[r] - mnew);
      mrow[r] = mnew;
    }
#pragma unroll
    for (int v = 0; v < 4; ++v)
      for (int r = 0; r < 8; ++r) acc[v][r] *= alpha[r];
#pragma unroll
    for (int r = 0; r < 8; ++r) {
      float s0 = 0.f;
#pragma unroll
      for (int j = 0; j < 4; ++j) {
        p[j][r] = __expf(p[j][r] - mrow[r]);
        s0 += p[j][r];
      }
      for (int off = 1; off < 16; off <<= 1) s0 += __shfl_xor(s0, off, 32);
      lrow[r] = lrow[r] * alpha[r] + s0;
    }

    // ---- relayout P (C-fragment layout) -> two A fragments via LDS ----
#pragma unroll
    for (int j = 0; j < 4; ++j)
      for (int r = 0; r < 8; ++r)
        plds[wave][r + laneHi * 8][j * 16 + ln] = (bf16_t)p[j][r];
    // wave-private region + in-order LDS pipe: a wave-local wait suffices
    asm volatile("s_wait_dscnt 0x0" ::: "memory");

    FragU f0, f1;
    const bf16_t* prow = &plds[wave][ln][0];
    f0.u[0] = *(const uint4*)(prow + laneHi * 8);        // k = 0..31 frag
    f0.u[1] = *(const uint4*)(prow + laneHi * 8 + 16);
    f1.u[0] = *(const uint4*)(prow + 32 + laneHi * 8);   // k = 32..63 frag
    f1.u[1] = *(const uint4*)(prow + 32 + laneHi * 8 + 16);
    asm volatile("s_wait_dscnt 0x0" ::: "memory");

    // ---- PV: acc[v] += P(16x64) x V(64x16) per DH tile ----
#pragma unroll
    for (int v = 0; v < 4; ++v) {
      acc[v] = wmma_bf16(f0.v, bv[v][0], acc[v]);
      acc[v] = wmma_bf16(f1.v, bv[v][1], acc[v]);
    }
  }

  // normalize and scatter to [B,T,D] (column = h*64 + dh)
  for (int v = 0; v < 4; ++v)
    for (int r = 0; r < 8; ++r) {
      int t = t0 + r + laneHi * 8;
      int d = h * DH_ + v * 16 + ln;
      O[((size_t)b * T_ + t) * D_ + d] = (bf16_t)(acc[v][r] / lrow[r]);
    }
}

// ---------------------------------------------------------------------------
extern "C" void kernel_launch(void* const* d_in, const int* in_sizes, int n_in,
                              void* d_out, int out_size, void* d_ws, size_t ws_size,
                              hipStream_t stream) {
  const float* query = (const float*)d_in[0];
  const float* key   = (const float*)d_in[1];
  const float* value = (const float*)d_in[2];
  const float* Wq = (const float*)d_in[3];
  const float* bq = (const float*)d_in[4];
  const float* Wk = (const float*)d_in[5];
  const float* bk = (const float*)d_in[6];
  const float* Wv = (const float*)d_in[7];
  const float* bv = (const float*)d_in[8];
  const float* Wo = (const float*)d_in[9];
  const float* bo = (const float*)d_in[10];
  float* out = (float*)d_out;

  const size_t MT  = (size_t)B_ * T_;      // 8192
  const size_t ACT = MT * D_;              // 8388608 elems
  const size_t WW  = (size_t)D_ * D_;      // 1048576 elems

  bf16_t* p = (bf16_t*)d_ws;
  bf16_t* q_bf  = p; p += ACT;
  bf16_t* k_bf  = p; p += ACT;
  bf16_t* v_bf  = p; p += ACT;
  bf16_t* wq_bf = p; p += WW;
  bf16_t* wk_bf = p; p += WW;
  bf16_t* wv_bf = p; p += WW;
  bf16_t* wo_bf = p; p += WW;
  bf16_t* Qh    = p; p += ACT;             // [B,H,T,DH]
  bf16_t* Kh    = p; p += ACT;             // [B,H,T,DH]
  bf16_t* Vt    = p; p += ACT;             // [B,H,DH,T]
  bf16_t* Abf   = p; p += ACT;             // attention out, [B,T,D]

  // f32 -> bf16 conversions
  {
    dim3 blk(256);
    dim3 ga((unsigned)((ACT + 255) / 256 > 4096 ? 4096 : (ACT + 255) / 256));
    dim3 gw((unsigned)((WW + 255) / 256));
    cvt_f32_bf16_kernel<<<ga, blk, 0, stream>>>(query, q_bf, (int)ACT);
    cvt_f32_bf16_kernel<<<ga, blk, 0, stream>>>(key,   k_bf, (int)ACT);
    cvt_f32_bf16_kernel<<<ga, blk, 0, stream>>>(value, v_bf, (int)ACT);
    cvt_f32_bf16_kernel<<<gw, blk, 0, stream>>>(Wq, wq_bf, (int)WW);
    cvt_f32_bf16_kernel<<<gw, blk, 0, stream>>>(Wk, wk_bf, (int)WW);
    cvt_f32_bf16_kernel<<<gw, blk, 0, stream>>>(Wv, wv_bf, (int)WW);
    cvt_f32_bf16_kernel<<<gw, blk, 0, stream>>>(Wo, wo_bf, (int)WW);
  }

  // projections: [8192,1024] x [1024,1024]^T ; block tile 64x128
  dim3 gg((unsigned)(MT / 64), (unsigned)(D_ / 128));
  gemm_xwT_kernel<<<gg, 128, 0, stream>>>(q_bf, wq_bf, bq, Qh, nullptr,
                                          (int)MT, D_, D_, 1);
  gemm_xwT_kernel<<<gg, 128, 0, stream>>>(k_bf, wk_bf, bk, Kh, nullptr,
                                          (int)MT, D_, D_, 1);
  gemm_xwT_kernel<<<gg, 128, 0, stream>>>(v_bf, wv_bf, bv, Vt, nullptr,
                                          (int)MT, D_, D_, 2);

  // flash attention
  dim3 gatt((unsigned)(T_ / 64), H_, B_);
  attn_kernel<<<gatt, 128, 0, stream>>>(Qh, Kh, Vt, Abf);

  // output projection -> f32 d_out
  gemm_xwT_kernel<<<gg, 128, 0, stream>>>(Abf, wo_bf, bo, nullptr, out,
                                          (int)MT, D_, D_, 3);
}
